// TransformerModel_84516366451050
// MI455X (gfx1250) — matrix operand order repaired and checked by reading, not compiled, not run on hardware
//
#include <hip/hip_runtime.h>
#include <hip/hip_bf16.h>
#include <math.h>

// ---------------------------------------------------------------------------
// MI455X (gfx1250) transformer forward.
// GEMMs: V_WMMA_F32_16X16X32_BF16; both A and B tiles are bf16 in memory and
// staged into LDS by the Tensor Data Mover (tensor_load_to_lds with LDS pad,
// s_wait_tensorcnt), double-buffered. Waves 0/1 drive the TDM, all 8 compute.
// ---------------------------------------------------------------------------

typedef __attribute__((ext_vector_type(8)))  __bf16 bf16x8;
typedef __attribute__((ext_vector_type(16))) __bf16 bf16x16;
typedef __attribute__((ext_vector_type(8)))  float  f32x8;
typedef __attribute__((ext_vector_type(4)))  unsigned int u32x4;
typedef __attribute__((ext_vector_type(8)))  int    i32x8;
typedef __attribute__((ext_vector_type(4)))  int    i32x4;

#define BM 64
#define BN 128
#define BK 32
#define KPAD 8   // LDS row = BK+8 = 40 elems (80B, 16B-aligned, conflict-free)

#if defined(__has_builtin)
#if __has_builtin(__builtin_amdgcn_tensor_load_to_lds) && __has_builtin(__builtin_amdgcn_s_wait_tensorcnt)
#define HAVE_TDM 1
#endif
#endif

#ifdef HAVE_TDM
// TDM 2D tile load: src[rows][32] bf16 slice of a row-major [*, stride] tensor
// -> LDS [rows][32+8] (D# pad: 4 dwords appended after every 16 dwords).
__device__ inline void tdm_load_tile(const __bf16* gsrc, unsigned lds_off,
                                     unsigned rows, unsigned valid_rows,
                                     unsigned stride_elems)
{
    unsigned long long ga = (unsigned long long)(uintptr_t)gsrc;
    u32x4 g0;
    g0[0] = 1u;                                        // count=1 (valid user D#)
    g0[1] = lds_off;                                   // lds_addr
    g0[2] = (unsigned)(ga & 0xffffffffu);              // global_addr[31:0]
    g0[3] = (unsigned)((ga >> 32) & 0x01ffffffu) | (2u << 30); // addr[56:32], type=2

    unsigned td0 = 32u;                                // tensor_dim0 (K elems)
    unsigned td1 = valid_rows;                         // OOB rows read as zero
    unsigned long long s0 = stride_elems;              // tensor_dim0_stride

    i32x8 g1;
    // data_size=1 (2B) | pad_enable | pad_interval=3 (16 dw) | pad_amount=3 (4 dw)
    g1[0] = (int)((1u << 16) | (1u << 20) | (3u << 22) | (3u << 25));
    g1[1] = (int)((td0 & 0xffffu) << 16);
    g1[2] = (int)((td0 >> 16) | ((td1 & 0xffffu) << 16));
    g1[3] = (int)((td1 >> 16) | (32u << 16));          // tile_dim0 = 32
    g1[4] = (int)(rows & 0xffffu);                     // tile_dim1, tile_dim2=0
    g1[5] = (int)(s0 & 0xffffffffu);
    g1[6] = (int)((s0 >> 32) & 0xffffu);               // stride1 = 0
    g1[7] = 0;

    i32x4 z4 = {0, 0, 0, 0};
    i32x8 z8 = {0, 0, 0, 0, 0, 0, 0, 0};
    __builtin_amdgcn_tensor_load_to_lds(g0, g1, z4, z4, z8, 0);
}
#endif

// ---------------------------------------------------------------------------
// GEMM: act(A[M,K] @ W[K,N] + bias) -> C (f32, strided/col-offset) and/or
// C16 (bf16). A is bf16 [M,K]; W is pre-transposed bf16 [N,K].
// 256 threads = 8 waves; wave (wm 0..3, wn 0..1) owns a 16x64 strip.
// ---------------------------------------------------------------------------
template<int ACT>
__global__ __launch_bounds__(256)
void gemm_wmma_bf16(const __bf16* __restrict__ A, int lda,
                    const __bf16* __restrict__ Wt,
                    const float* __restrict__ bias,
                    float* __restrict__ C, int ldc, int cofs,
                    __bf16* __restrict__ C16, int ldc16,
                    int M, int N, int K)
{
    __shared__ alignas(16) __bf16 As[2][BM][BK + KPAD];
    __shared__ alignas(16) __bf16 Bs[2][BN][BK + KPAD];

    const int tid  = threadIdx.x;
    const int lane = tid & 31;
    const int wave = tid >> 5;
    const int wm   = wave & 3;
    const int wn   = wave >> 2;
    const int tileM = blockIdx.y * BM;
    const int tileN = blockIdx.x * BN;
    const int ncols = (N - tileN < BN) ? (N - tileN) : BN;

    f32x8 acc[4] = {};

    auto stage = [&](int buf, int k0) {
#ifdef HAVE_TDM
        if (wave == 0) {
            unsigned lds = (unsigned)(unsigned long long)(uintptr_t)&As[buf][0][0];
            tdm_load_tile(A + (size_t)tileM * lda + k0, lds, BM, BM, (unsigned)lda);
        } else if (wave == 1) {
            unsigned lds = (unsigned)(unsigned long long)(uintptr_t)&Bs[buf][0][0];
            tdm_load_tile(Wt + (size_t)tileN * K + k0, lds, BN, (unsigned)ncols, (unsigned)K);
        }
#else
        {   // A: 1 bf16x8 copy per thread (64 rows x 4 k-chunks).
            int row = tid >> 2, kc = tid & 3;
            *(bf16x8*)&As[buf][row][kc * 8] =
                *(const bf16x8*)&A[(size_t)(tileM + row) * lda + k0 + kc * 8];
        }
        #pragma unroll
        for (int i = 0; i < 2; ++i) {   // B: 2 bf16x8 copies per thread.
            int task = i * 256 + tid;
            int kc = task & 3, col = task >> 2;
            bf16x8 p = {};
            if (col < ncols)
                p = *(const bf16x8*)&Wt[(size_t)(tileN + col) * K + k0 + kc * 8];
            *(bf16x8*)&Bs[buf][col][kc * 8] = p;
        }
#endif
    };
    auto wait_stage = [&]() {
#ifdef HAVE_TDM
        if (wave < 2) __builtin_amdgcn_s_wait_tensorcnt(0);
#endif
    };

    stage(0, 0);
    wait_stage();
    __syncthreads();

    const int nk = K / BK;
    for (int kt = 0; kt < nk; ++kt) {
        const int buf = kt & 1;
        if (kt + 1 < nk) stage(buf ^ 1, (kt + 1) * BK);

        // Load ALL fragments first (10x ds_load_b128), then issue 4 WMMAs
        // back-to-back so one dscnt wait covers the whole group.
        const int row = wm * 16 + (lane & 15);
        const int akb = (lane >> 4) * 8;
        bf16x8 alo = *(const bf16x8*)&As[buf][row][akb];
        bf16x8 ahi = *(const bf16x8*)&As[buf][row][akb + 16];

        const int bkb = (lane >> 4) * 8;
        bf16x8 bl[4], bh[4];
        #pragma unroll
        for (int t = 0; t < 4; ++t) {
            int col = wn * 64 + t * 16 + (lane & 15);
            bl[t] = *(const bf16x8*)&Bs[buf][col][bkb];
            bh[t] = *(const bf16x8*)&Bs[buf][col][bkb + 16];
        }

        bf16x16 af = __builtin_shufflevector(alo, ahi,
             0, 1, 2, 3, 4, 5, 6, 7, 8, 9, 10, 11, 12, 13, 14, 15);
        #pragma unroll
        for (int t = 0; t < 4; ++t) {
            bf16x16 bfv = __builtin_shufflevector(bl[t], bh[t],
                 0, 1, 2, 3, 4, 5, 6, 7, 8, 9, 10, 11, 12, 13, 14, 15);
            acc[t] = __builtin_amdgcn_wmma_f32_16x16x32_bf16(
                false, af, false, bfv, (short)0, acc[t], false, false);
        }

        if (kt + 1 < nk) wait_stage();
        __syncthreads();
    }

    // Epilogue. C/D layout: VGPR r, lane l -> M = r + 8*(l>=16), N = l&15.
    #pragma unroll
    for (int t = 0; t < 4; ++t) {
        int col = tileN + wn * 64 + t * 16 + (lane & 15);
        if (col < N) {
            float bv = bias ? bias[col] : 0.0f;
            #pragma unroll
            for (int r = 0; r < 8; ++r) {
                int row = tileM + wm * 16 + r + ((lane >> 4) << 3);
                float xv = acc[t][r] + bv;
                if (ACT == 1)
                    xv = 0.5f * xv * (1.0f + erff(xv * 0.70710678118654752f));
                if (C)   C[(size_t)row * ldc + cofs + col] = xv;
                if (C16) C16[(size_t)row * ldc16 + col] = (__bf16)xv;
            }
        }
    }
}

// ---------------------------------------------------------------------------
// Weight convert+transpose prepass: W fp32 [batch,K,N] -> Wt bf16 [batch,N,K].
// ---------------------------------------------------------------------------
__global__ void wcvt_transpose(const float* __restrict__ W, __bf16* __restrict__ Wt,
                               int K, int N, long total)
{
    long idx = blockIdx.x * (long)blockDim.x + threadIdx.x;
    if (idx >= total) return;
    long per = (long)K * N;
    long b   = idx / per;
    long rem = idx - b * per;
    int  n   = (int)(rem / K);
    int  kk  = (int)(rem - (long)n * K);
    Wt[idx] = (__bf16)W[b * per + (size_t)kk * N + n];
}

// ---------------------------------------------------------------------------
// Multi-field embedding gather -> bf16 [tok, 1280]
// ---------------------------------------------------------------------------
__global__ void embed_kernel(const int* __restrict__ x,
                             const float* __restrict__ e0, const float* __restrict__ e1,
                             const float* __restrict__ e2, const float* __restrict__ e3,
                             const float* __restrict__ e4, const float* __restrict__ e5,
                             const float* __restrict__ e6, const float* __restrict__ e7,
                             __bf16* __restrict__ out16, int total)
{
    int idx = blockIdx.x * blockDim.x + threadIdx.x;
    if (idx >= total) return;
    int tok = idx / 1280;
    int col = idx - tok * 1280;
    int f, base, width;
    const float* tab;
    if      (col <   32) { f = 0; base = 0;    width = 32;  tab = e0; }
    else if (col <  160) { f = 1; base = 32;   width = 128; tab = e1; }
    else if (col <  416) { f = 2; base = 160;  width = 256; tab = e2; }
    else if (col <  480) { f = 3; base = 416;  width = 64;  tab = e3; }
    else if (col <  512) { f = 4; base = 480;  width = 32;  tab = e4; }
    else if (col < 1024) { f = 5; base = 512;  width = 512; tab = e5; }
    else if (col < 1152) { f = 6; base = 1024; width = 128; tab = e6; }
    else                 { f = 7; base = 1152; width = 128; tab = e7; }
    int id = x[tok * 8 + f];
    out16[idx] = (__bf16)(tab[(size_t)id * width + (col - base)] * sqrtf((float)width));
}

// ---------------------------------------------------------------------------
// Sinusoidal positional encoding add; writes f32 (for residual) + bf16 (GEMM A).
// ---------------------------------------------------------------------------
__global__ void pe_add_kernel(float* __restrict__ h, __bf16* __restrict__ h16, int total)
{
    int idx = blockIdx.x * blockDim.x + threadIdx.x;
    if (idx >= total) return;
    int tok = idx >> 9;
    int col = idx & 511;
    float pos = (float)(tok & 2047);
    float i2  = (float)(col & ~1);
    float ang = pos * __expf(-i2 * (9.2103403719761836f / 512.0f));
    float vv  = h[idx] + ((col & 1) ? __cosf(ang) : __sinf(ang));
    h[idx]   = vv;
    h16[idx] = (__bf16)vv;
}

// ---------------------------------------------------------------------------
// Fused residual + LayerNorm over D=512. out (f32) and out16 (bf16) nullable.
// ---------------------------------------------------------------------------
__global__ __launch_bounds__(256)
void ln_kernel(const float* __restrict__ xa, const float* __restrict__ xb,
               const float* __restrict__ g, const float* __restrict__ b,
               float* __restrict__ out, __bf16* __restrict__ out16)
{
    __shared__ float s1[256], s2[256];
    const int row = blockIdx.x;
    const int t   = threadIdx.x;
    size_t base = (size_t)row * 512;
    float v0 = xa[base + t];
    float v1 = xa[base + 256 + t];
    if (xb) { v0 += xb[base + t]; v1 += xb[base + 256 + t]; }
    s1[t] = v0 + v1;
    s2[t] = v0 * v0 + v1 * v1;
    __syncthreads();
    for (int w = 128; w > 0; w >>= 1) {
        if (t < w) { s1[t] += s1[t + w]; s2[t] += s2[t + w]; }
        __syncthreads();
    }
    float mean = s1[0] * (1.0f / 512.0f);
    float var  = s2[0] * (1.0f / 512.0f) - mean * mean;
    float rstd = rsqrtf(var + 1e-5f);
    float r0 = (v0 - mean) * rstd * g[t]       + b[t];
    float r1 = (v1 - mean) * rstd * g[256 + t] + b[256 + t];
    if (out)   { out[base + t] = r0;           out[base + 256 + t] = r1; }
    if (out16) { out16[base + t] = (__bf16)r0; out16[base + 256 + t] = (__bf16)r1; }
}

// ---------------------------------------------------------------------------
// Causal linear attention scan (phi(x)=elu(x)+1, inclusive cumsum) -> bf16 out.
// ---------------------------------------------------------------------------
__global__ __launch_bounds__(64)
void linattn_scan(const float* __restrict__ q, const float* __restrict__ k,
                  const float* __restrict__ v, __bf16* __restrict__ o16)
{
    const int bh = blockIdx.x;
    const int b  = bh >> 3;
    const int h  = bh & 7;
    const int t  = threadIdx.x;

    __shared__ float qs[64], ks[64], red[64], zsh;

    float st[64];
    #pragma unroll
    for (int i = 0; i < 64; ++i) st[i] = 0.0f;
    float ksum = 0.0f;

    size_t base = ((size_t)b * 2048) * 512 + h * 64;
    for (int s = 0; s < 2048; ++s) {
        size_t off = base + (size_t)s * 512 + t;
        float qr = q[off], kr = k[off], vown = v[off];
        float qp = (qr > 0.0f) ? (qr + 1.0f) : __expf(qr);
        float kp = (kr > 0.0f) ? (kr + 1.0f) : __expf(kr);
        ksum += kp;
        qs[t]  = qp;
        ks[t]  = kp;
        red[t] = qp * ksum;
        __syncthreads();
        for (int w = 32; w > 0; w >>= 1) {
            if (t < w) red[t] += red[t + w];
            __syncthreads();
        }
        if (t == 0) zsh = 1.0f / (red[0] + 1e-6f);
        __syncthreads();
        float acc = 0.0f;
        #pragma unroll
        for (int d = 0; d < 64; ++d) {
            st[d] += ks[d] * vown;
            acc   += qs[d] * st[d];
        }
        o16[off] = (__bf16)(acc * zsh);
        __syncthreads();
    }
}

// ---------------------------------------------------------------------------
// Concat [hf16 (512) | emb_type[target[:,4]] * sqrt(32) (32)] -> cc16[tok,544]
// ---------------------------------------------------------------------------
__global__ void concat_kernel(const __bf16* __restrict__ hf16,
                              const int* __restrict__ target,
                              const float* __restrict__ emb_type,
                              __bf16* __restrict__ cc16, int total)
{
    int idx = blockIdx.x * blockDim.x + threadIdx.x;
    if (idx >= total) return;
    int tok = idx / 544;
    int col = idx - tok * 544;
    if (col < 512) {
        cc16[idx] = hf16[(size_t)tok * 512 + col];
    } else {
        int id = target[tok * 8 + 4];
        cc16[idx] = (__bf16)(emb_type[id * 32 + (col - 512)] * 5.6568542494923806f);
    }
}

// ---------------------------------------------------------------------------
// Host-side orchestration.
// ---------------------------------------------------------------------------
static inline void launch_gemm(int act,
                               const __bf16* A, int lda, const __bf16* Wt,
                               const float* bias,
                               float* C, int ldc, int cofs,
                               __bf16* C16, int ldc16,
                               int M, int N, int K, hipStream_t stream)
{
    dim3 grid((N + BN - 1) / BN, M / BM);
    if (act == 1)
        gemm_wmma_bf16<1><<<grid, 256, 0, stream>>>(A, lda, Wt, bias, C, ldc, cofs, C16, ldc16, M, N, K);
    else
        gemm_wmma_bf16<0><<<grid, 256, 0, stream>>>(A, lda, Wt, bias, C, ldc, cofs, C16, ldc16, M, N, K);
}

static inline void launch_wcvt(const float* W, __bf16* Wt, int K, int N, int batch,
                               hipStream_t stream)
{
    long total = (long)batch * K * N;
    wcvt_transpose<<<(unsigned)((total + 255) / 256), 256, 0, stream>>>(W, Wt, K, N, total);
}

extern "C" void kernel_launch(void* const* d_in, const int* in_sizes, int n_in,
                              void* d_out, int out_size, void* d_ws, size_t ws_size,
                              hipStream_t stream)
{
    const int M  = 4096;   // B*S
    const int D  = 512;
    const int FF = 2048;
    const int L  = 12;

    const int*   x      = (const int*)d_in[0];
    const int*   target = (const int*)d_in[1];
    const float* emb[8];
    for (int i = 0; i < 8; ++i) emb[i] = (const float*)d_in[2 + i];
    const float* in_w  = (const float*)d_in[10];
    const float* in_b  = (const float*)d_in[11];
    const float* wq    = (const float*)d_in[12];
    const float* wk    = (const float*)d_in[13];
    const float* wv    = (const float*)d_in[14];
    const float* wo    = (const float*)d_in[15];
    const float* bq    = (const float*)d_in[16];
    const float* bk    = (const float*)d_in[17];
    const float* bv    = (const float*)d_in[18];
    const float* bo    = (const float*)d_in[19];
    const float* ln1_g = (const float*)d_in[20];
    const float* ln1_b = (const float*)d_in[21];
    const float* ln2_g = (const float*)d_in[22];
    const float* ln2_b = (const float*)d_in[23];
    const float* w1    = (const float*)d_in[24];
    const float* b1    = (const float*)d_in[25];
    const float* w2    = (const float*)d_in[26];
    const float* b2    = (const float*)d_in[27];
    const float* lnf_g = (const float*)d_in[28];
    const float* lnf_b = (const float*)d_in[29];
    const float* pc_w  = (const float*)d_in[30];
    const float* pc_b  = (const float*)d_in[31];

    static const int NTv[8]  = {4, 56, 135, 18, 4, 87, 18, 25};
    static const int OFSv[8] = {0, 4, 60, 195, 213, 217, 304, 322};

    // ---- workspace carve-up ----
    char*  wsb = (char*)d_ws;
    size_t off = 0;
    auto allocf = [&](size_t n) { float* p = (float*)(wsb + off); off += n * 4; return p; };
    auto allocb = [&](size_t n) { __bf16* p = (__bf16*)(wsb + off);
                                  off += ((n * 2 + 15) & ~(size_t)15); return p; };

    // f32 activations (residual stream + scan inputs)
    float* h    = allocf((size_t)M * D);
    float* qb   = allocf((size_t)M * D);
    float* kbuf = allocf((size_t)M * D);
    float* vbuf = allocf((size_t)M * D);
    float* abuf = allocf((size_t)M * D);
    float* xr   = allocf((size_t)M * D);
    float* yb   = allocf((size_t)M * D);
    // bf16 activations (GEMM A operands)
    __bf16* embcat16 = allocb((size_t)M * 1280);
    __bf16* h16      = allocb((size_t)M * D);
    __bf16* o16      = allocb((size_t)M * D);
    __bf16* xr16     = allocb((size_t)M * D);
    __bf16* ff16     = allocb((size_t)M * FF);
    __bf16* hf16     = allocb((size_t)M * D);
    __bf16* cc16     = allocb((size_t)M * 544);
    __bf16* y2_16    = allocb((size_t)M * D);
    // bf16 transposed weights [N,K]
    __bf16* in_w_t = allocb((size_t)1280 * D);
    __bf16* wq_t   = allocb((size_t)L * D * D);
    __bf16* wk_t   = allocb((size_t)L * D * D);
    __bf16* wv_t   = allocb((size_t)L * D * D);
    __bf16* wo_t   = allocb((size_t)L * D * D);
    __bf16* w1_t   = allocb((size_t)L * D * FF);
    __bf16* w2_t   = allocb((size_t)L * FF * D);
    __bf16* pc_w_t = allocb((size_t)544 * D);
    __bf16* pw_t[8];
    for (int i = 0; i < 8; ++i) pw_t[i] = allocb((size_t)D * NTv[i]);

    // ---- 0) weight convert + transpose prepass ----
    launch_wcvt(in_w, in_w_t, 1280, D, 1, stream);
    launch_wcvt(wq, wq_t, D, D, L, stream);
    launch_wcvt(wk, wk_t, D, D, L, stream);
    launch_wcvt(wv, wv_t, D, D, L, stream);
    launch_wcvt(wo, wo_t, D, D, L, stream);
    launch_wcvt(w1, w1_t, D, FF, L, stream);
    launch_wcvt(w2, w2_t, FF, D, L, stream);
    launch_wcvt(pc_w, pc_w_t, 544, D, 1, stream);
    for (int i = 0; i < 8; ++i)
        launch_wcvt((const float*)d_in[32 + 2 * i], pw_t[i], D, NTv[i], 1, stream);

    // ---- 1) embeddings -> embcat16 ----
    {
        int total = M * 1280;
        embed_kernel<<<(total + 255) / 256, 256, 0, stream>>>(
            x, emb[0], emb[1], emb[2], emb[3], emb[4], emb[5], emb[6], emb[7],
            embcat16, total);
    }
    // ---- 2) input projection (f32 out) + 3) PE (f32 + bf16 out) ----
    launch_gemm(0, embcat16, 1280, in_w_t, in_b, h, D, 0, nullptr, 0, M, D, 1280, stream);
    pe_add_kernel<<<(M * D + 255) / 256, 256, 0, stream>>>(h, h16, M * D);

    // ---- 4) transformer layers ----
    for (int l = 0; l < L; ++l) {
        size_t dd = (size_t)l * D * D;
        launch_gemm(0, h16, D, wq_t + dd, bq + (size_t)l * D, qb,   D, 0, nullptr, 0, M, D, D, stream);
        launch_gemm(0, h16, D, wk_t + dd, bk + (size_t)l * D, kbuf, D, 0, nullptr, 0, M, D, D, stream);
        launch_gemm(0, h16, D, wv_t + dd, bv + (size_t)l * D, vbuf, D, 0, nullptr, 0, M, D, D, stream);
        linattn_scan<<<16, 64, 0, stream>>>(qb, kbuf, vbuf, o16);
        launch_gemm(0, o16, D, wo_t + dd, bo + (size_t)l * D, abuf, D, 0, nullptr, 0, M, D, D, stream);
        ln_kernel<<<M, 256, 0, stream>>>(h, abuf, ln1_g + (size_t)l * D, ln1_b + (size_t)l * D, xr, xr16);
        launch_gemm(1, xr16, D, w1_t + (size_t)l * D * FF, b1 + (size_t)l * FF,
                    nullptr, 0, 0, ff16, FF, M, FF, D, stream);
        launch_gemm(0, ff16, FF, w2_t + (size_t)l * FF * D, b2 + (size_t)l * D,
                    yb, D, 0, nullptr, 0, M, D, FF, stream);
        ln_kernel<<<M, 256, 0, stream>>>(xr, yb, ln2_g + (size_t)l * D, ln2_b + (size_t)l * D, h, h16);
    }

    // ---- 5) final LN (bf16 only) ----
    ln_kernel<<<M, 256, 0, stream>>>(h, nullptr, lnf_g, lnf_b, nullptr, hf16);

    float* out = (float*)d_out;   // [M, 347]

    // ---- 6) type head from hf16 ----
    launch_gemm(0, hf16, D, pw_t[4], (const float*)d_in[33 + 2 * 4],
                out, 347, OFSv[4], nullptr, 0, M, NTv[4], D, stream);
    // ---- 7) concat + projection (bf16 only) ----
    {
        int total = M * 544;
        concat_kernel<<<(total + 255) / 256, 256, 0, stream>>>(hf16, target, emb[4], cc16, total);
    }
    launch_gemm(0, cc16, 544, pc_w_t, pc_b, nullptr, 0, 0, y2_16, D, M, D, 544, stream);
    // ---- 8) remaining heads from y2_16 ----
    for (int i = 0; i < 8; ++i) {
        if (i == 4) continue;
        launch_gemm(0, y2_16, D, pw_t[i], (const float*)d_in[33 + 2 * i],
                    out, 347, OFSv[i], nullptr, 0, M, NTv[i], D, stream);
    }
    (void)in_sizes; (void)n_in; (void)out_size; (void)ws_size;
}
